// InformationAggregationLayer_74217034874925
// MI455X (gfx1250) — compile-verified, matrix operand order checked by reading, not run on hardware
//
#include <hip/hip_runtime.h>
#include <stdint.h>

// ---------------------------------------------------------------------------
// InformationAggregationLayer, fused for MI455X (gfx1250, wave32, WMMA).
// Bandwidth-bound: hist = 335 MB f32 read once; h_proj/gate never hit HBM.
// Matrix math via v_wmma_f32_16x16x32_bf16 with f32 accumulation.
// Round 4: TDM double-buffered pipeline, fixed for the 6-arg
// tensor_load_to_lds builtin and constant-immediate s_wait_tensorcnt.
// ---------------------------------------------------------------------------

typedef __attribute__((ext_vector_type(16))) __bf16 v16bf;
typedef __attribute__((ext_vector_type(8)))  float  v8f;
typedef __attribute__((ext_vector_type(4)))  unsigned int u32x4;
typedef __attribute__((ext_vector_type(4)))  int i32x4;
typedef __attribute__((ext_vector_type(8)))  int i32x8;

#define NB 8
#define NH 20
#define NS 1024
#define NE 256
#define D2 512   // 2*E

// Fused-kernel LDS layout (bytes). Total ~289 KB < 320 KB/WGP; only 1 WG/WGP.
#define OFF_A     0                        // 32*512 bf16 swizzled   (32 KB)
#define OFF_SUM   (32 * D2 * 2)            // 32*512 f32 summed      (64 KB)
#define OFF_CW    (OFF_SUM + 32 * D2 * 4)  // 32*512 f32 c_proj+wh_b (64 KB)
#define OFF_AL    (OFF_CW + 32 * D2 * 4)   // 32 f32 alpha
#define OFF_RAW0  (OFF_AL + 1024)          // 32*512 f32 raw tile    (64 KB)
#define OFF_RAW1  (OFF_RAW0 + 32 * D2 * 4) // 32*512 f32 raw tile    (64 KB)
#define SMEM_TOTAL (OFF_RAW1 + 32 * D2 * 4)

#if defined(__HIP_DEVICE_COMPILE__) &&                         \
    __has_builtin(__builtin_amdgcn_tensor_load_to_lds) &&      \
    __has_builtin(__builtin_amdgcn_s_wait_tensorcnt)
#define USE_TDM 1
#else
#define USE_TDM 0
#endif

__device__ __forceinline__ unsigned short f2bf(float f) {
  unsigned int u = __float_as_uint(f);
  u += 0x7FFFu + ((u >> 16) & 1u);           // round-to-nearest-even
  return (unsigned short)(u >> 16);
}
__device__ __forceinline__ float bf2f(unsigned short h) {
  return __uint_as_float(((unsigned int)h) << 16);
}

// WMMA 16-bit operand layout (ISA 7.12.2): element (row r within 16, k within
// 32) lives at lane = r + 16*laneHi, halfword i, where
//   laneHi = (k>>3)&1 ;  i = (k&7) + 8*(k>=16)
// Fragment = 512 bf16 = 32 lanes x 16 halves, stored contiguously.
__device__ __forceinline__ int frag_off(int tile, int kl, int rowl) {
  int laneHi = (kl >> 3) & 1;
  int i = (kl & 7) + ((kl & 16) ? 8 : 0);
  int lane = rowl + (laneHi << 4);
  return (tile << 9) + (lane << 4) + i;
}

__device__ __forceinline__ v8f wmma_bf16(v16bf a, v16bf b, v8f c) {
  return __builtin_amdgcn_wmma_f32_16x16x32_bf16(false, a, false, b,
                                                 (short)0, c, false, false);
}

#if USE_TDM
// Issue a TDM DMA of one contiguous 32x512-f32 (64 KB) tile: global -> LDS.
// D# per CDNA5 ISA sec.8: group0 = {count=1, lds_addr, global_addr, type=2};
// group1 = {data_size=4B, tensor_dim0=tile_dim0=16384, dim1=1, strides}.
// Groups 2/3 unused (<=2-D tensor) -> zero-filled. Tracked on TENSORcnt;
// EXEC ignored, so call it from a single wave (one DMA per workgroup).
__device__ __forceinline__ void tdm_load_tile(const float* gsrc,
                                              unsigned ldsByteAddr) {
  unsigned long long ga = (unsigned long long)(uintptr_t)gsrc;
  u32x4 g0;
  g0.x = 1u;                                        // count=1 (valid D#)
  g0.y = ldsByteAddr;                               // lds_addr [63:32]
  g0.z = (unsigned)(ga & 0xffffffffu);              // global_addr lo
  g0.w = (unsigned)((ga >> 32) & 0x01ffffffu)       // global_addr[56:32]
         | (2u << 30);                              // type=2 ("image")
  i32x8 g1;
  g1.s0 = 0x00020000;    // wg_mask=0 | data_size=2 (4B) | no flags
  g1.s1 = 0x40000000;    // tensor_dim0[15:0]=0x4000 @bits[63:48]
  g1.s2 = 0x00010000;    // tensor_dim0 hi=0 | tensor_dim1=1 @bits[95:80]
  g1.s3 = 0x40000000;    // tensor_dim1 hi=0 | tile_dim0=0x4000 @bits[127:112]
  g1.s4 = 0x00000001;    // tile_dim1=1 | tile_dim2=0
  g1.s5 = 0x00004000;    // tensor_dim0_stride lo = 16384
  g1.s6 = 0x40000000;    // stride0 hi=0 | tensor_dim1_stride lo16 = 0x4000
  g1.s7 = 0;             // tensor_dim1_stride hi = 0
  i32x4 gz4 = {0, 0, 0, 0};
  i32x8 gz8 = {0, 0, 0, 0, 0, 0, 0, 0};
  __builtin_amdgcn_tensor_load_to_lds(g0, g1, gz4, gz4, gz8, 0);
}
#endif

// ---------------------------------------------------------------------------
// Kernel 0: convert a row-major f32 weight [N,K] into bf16 WMMA-B fragments.
// One thread handles 4 consecutive k (stays within one laneHi/i-contiguous
// group since 4 | 8), so the store is a single b64.
// ---------------------------------------------------------------------------
__global__ void __launch_bounds__(256, 1) w_swizzle(
    const float* __restrict__ W, unsigned short* __restrict__ out,
    int N, int K) {
  int t = blockIdx.x * 256 + threadIdx.x;
  int k = (t << 2) % K;
  int n = (t << 2) / K;
  if (n >= N) return;
  float4 v = *(const float4*)(W + (size_t)n * K + k);
  int NKT = K >> 5;
  int off = frag_off((n >> 4) * NKT + (k >> 5), k & 31, n & 15);
  ushort4 q;
  q.x = f2bf(v.x); q.y = f2bf(v.y); q.z = f2bf(v.z); q.w = f2bf(v.w);
  *(ushort4*)(out + off) = q;
}

// ---------------------------------------------------------------------------
// Generic WMMA GEMM:  D[32 x N] = A[32 x K] * Bsw^T + bias,  f32 out.
// A rows come from A1 (k < K1) then A2 (k >= K1)  -> lets the final GEMM read
// the logical concat [cur | summed] without materializing it.
// ---------------------------------------------------------------------------
template <int NTPER>
__global__ void __launch_bounds__(256, 1) gemm_bf16(
    const float* __restrict__ A1, const float* __restrict__ A2,
    int K1, int K,
    const unsigned short* __restrict__ Bsw,
    const float* __restrict__ bias,
    float* __restrict__ D, int N) {
  extern __shared__ char smem[];
  unsigned short* ldsA = (unsigned short*)smem;  // 32*K bf16, swizzled

  const int tid  = threadIdx.x;
  const int lane = tid & 31;
  const int wave = tid >> 5;
  const int row0 = blockIdx.x * 32;
  const int NKT  = K >> 5;
  const int K2   = K - K1;

  // Stage A tile into LDS as swizzled bf16.
  const int perT = K >> 3;  // (32*K)/256 elements per thread
  for (int j = 0; j < perT; j += 4) {
    int lin = tid * perT + j;
    int m = lin / K;
    int k = lin - m * K;
    const float* src = (k < K1) ? (A1 + (size_t)(row0 + m) * K1 + k)
                                : (A2 + (size_t)(row0 + m) * K2 + (k - K1));
    float4 v = *(const float4*)src;
    int off = frag_off((m >> 4) * NKT + (k >> 5), k & 31, m & 15);
    ushort4 q;
    q.x = f2bf(v.x); q.y = f2bf(v.y); q.z = f2bf(v.z); q.w = f2bf(v.w);
    *(ushort4*)(&ldsA[off]) = q;
  }
  __syncthreads();

  const int nt0 = wave * NTPER;
  const v8f vz = {0.f, 0.f, 0.f, 0.f, 0.f, 0.f, 0.f, 0.f};
  v8f acc[2][NTPER];
#pragma unroll
  for (int mt = 0; mt < 2; ++mt)
#pragma unroll
    for (int j = 0; j < NTPER; ++j) acc[mt][j] = vz;

#pragma unroll 2
  for (int kt = 0; kt < NKT; ++kt) {
    v16bf a0 = *(const v16bf*)(&ldsA[((0 * NKT + kt) << 9) + (lane << 4)]);
    v16bf a1 = *(const v16bf*)(&ldsA[((1 * NKT + kt) << 9) + (lane << 4)]);
#pragma unroll
    for (int j = 0; j < NTPER; ++j) {
      const unsigned short* bp =
          Bsw + (((size_t)(nt0 + j) * NKT + kt) << 9) + (lane << 4);
      v16bf bf = *(const v16bf*)bp;
      acc[0][j] = wmma_bf16(a0, bf, acc[0][j]);
      acc[1][j] = wmma_bf16(a1, bf, acc[1][j]);
    }
  }

  const int laneHi = lane >> 4;
  const int nl = lane & 15;
#pragma unroll
  for (int mt = 0; mt < 2; ++mt)
#pragma unroll
    for (int j = 0; j < NTPER; ++j) {
      int n = ((nt0 + j) << 4) + nl;
      float bv = bias[n];
#pragma unroll
      for (int i = 0; i < 8; ++i) {
        int row = row0 + (mt << 4) + i + (laneHi << 3);
        D[(size_t)row * N + n] = acc[mt][j][i] + bv;
      }
    }
}

// ---------------------------------------------------------------------------
// Fused kernel: per (b, 32-row s-tile), loop over h:
//   h_proj tile via WMMA  ->  gate=sigmoid(+c_proj+wh_b)  ->  alpha=qt.gate
//   ->  summed += alpha * hist  (LDS accumulator, hist tile reused from LDS)
// TDM path: next h tile is DMA'd (raw f32) into a double buffer while the
// current tile runs through WMMA, so the HBM stream overlaps compute.
// ---------------------------------------------------------------------------
__global__ void __launch_bounds__(256, 1) fused_attn(
    const float* __restrict__ hist,
    const float* __restrict__ c_proj,
    const unsigned short* __restrict__ whB,
    const float* __restrict__ wh_b,
    const float* __restrict__ qt_w,
    const float* __restrict__ qt_b,
    float* __restrict__ summed) {
  extern __shared__ char smem[];
  unsigned short* ldsA = (unsigned short*)(smem + OFF_A);
  float* ldsSum = (float*)(smem + OFF_SUM);
  float* ldsCW  = (float*)(smem + OFF_CW);
  float* alphaS = (float*)(smem + OFF_AL);

  const int tid    = threadIdx.x;
  const int lane   = tid & 31;
  const int wave   = tid >> 5;
  const int laneHi = lane >> 4;
  const int nl     = lane & 15;
  const int b      = blockIdx.x >> 5;
  const int s0     = (blockIdx.x & 31) << 5;
  const float qtb  = qt_b[0];
  const int nt0    = wave << 2;  // 4 N-tiles per wave, 8 waves cover N=512

  // qt_w per-lane values (only 4 VGPRs).
  float qn[4];
#pragma unroll
  for (int j = 0; j < 4; ++j) qn[j] = qt_w[((nt0 + j) << 4) + nl];

  // Stage (c_proj + wh_b) for this s-tile into LDS (constant across h), and
  // zero the summed accumulator. Each thread owns row m = tid/8, 64 columns.
  {
    int m = tid >> 3;
    int n0 = (tid & 7) << 6;
    const float* cp = c_proj + ((size_t)b * NS + s0 + m) * D2 + n0;
    float4 z = {0.f, 0.f, 0.f, 0.f};
    for (int dd = 0; dd < 64; dd += 4) {
      float4 c = *(const float4*)(cp + dd);
      float4 w = *(const float4*)(wh_b + n0 + dd);
      float4 r; r.x = c.x + w.x; r.y = c.y + w.y; r.z = c.z + w.z; r.w = c.w + w.w;
      *(float4*)(&ldsCW[(m << 9) + n0 + dd]) = r;
      *(float4*)(&ldsSum[(m << 9) + n0 + dd]) = z;
    }
  }

  const v8f vz = {0.f, 0.f, 0.f, 0.f, 0.f, 0.f, 0.f, 0.f};
  const size_t hStride = (size_t)NS * D2;
  const float* htBase = hist + ((size_t)b * NH * NS + s0) * D2;

#if USE_TDM
  const unsigned rawAddr0 = (unsigned)(uintptr_t)(smem + OFF_RAW0);
  const unsigned rawAddr1 = (unsigned)(uintptr_t)(smem + OFF_RAW1);
  if (tid == 0) tdm_load_tile(htBase, rawAddr0);  // prime the pipeline
#endif

  for (int h = 0; h < NH; ++h) {
    __syncthreads();  // prior-h LDS consumers done before refill
    const float* ht = htBase + (size_t)h * hStride;

#if USE_TDM
    if (tid == 0) {
      if (h + 1 < NH) {
        // kick DMA for next tile, then drain current one (in-order cnt)
        tdm_load_tile(ht + hStride, ((h + 1) & 1) ? rawAddr1 : rawAddr0);
        __builtin_amdgcn_s_wait_tensorcnt(1);
      } else {
        __builtin_amdgcn_s_wait_tensorcnt(0);
      }
    }
    __syncthreads();  // raw tile h visible to all waves
    const float* rawp = (const float*)(smem + ((h & 1) ? OFF_RAW1 : OFF_RAW0));
    for (int j = 0; j < 64; j += 4) {  // LDS f32 -> swizzled bf16 fragments
      int lin = (tid << 6) + j;
      float4 v = *(const float4*)(rawp + lin);
      int m = lin >> 9;
      int d = lin & (D2 - 1);
      int off = frag_off(((m >> 4) << 4) + (d >> 5), d & 31, m & 15);
      ushort4 q;
      q.x = f2bf(v.x); q.y = f2bf(v.y); q.z = f2bf(v.z); q.w = f2bf(v.w);
      *(ushort4*)(&ldsA[off]) = q;
    }
#else
    // Hist tile [32 x 512] is one contiguous 64 KB block -> coalesced b128.
    for (int j = 0; j < 64; j += 4) {
      int lin = (tid << 6) + j;
      float4 v = *(const float4*)(ht + lin);
      int m = lin >> 9;
      int d = lin & (D2 - 1);
      int off = frag_off(((m >> 4) << 4) + (d >> 5), d & 31, m & 15);
      ushort4 q;
      q.x = f2bf(v.x); q.y = f2bf(v.y); q.z = f2bf(v.z); q.w = f2bf(v.w);
      *(ushort4*)(&ldsA[off]) = q;
    }
    if (h + 1 < NH) {  // pull next tile toward cache (global_prefetch_b8)
      __builtin_prefetch(ht + hStride + (tid << 6), 0, 1);
      __builtin_prefetch(ht + hStride + (tid << 6) + 32, 0, 1);
    }
#endif
    if (tid < 32) alphaS[tid] = qtb;  // alpha starts at qt_b
    __syncthreads();

    // h_proj tile: 2 M-tiles x 4 N-tiles, K=512 in 16 steps of 32.
    v8f acc[2][4];
#pragma unroll
    for (int mt = 0; mt < 2; ++mt)
#pragma unroll
      for (int j = 0; j < 4; ++j) acc[mt][j] = vz;

#pragma unroll 2
    for (int kt = 0; kt < 16; ++kt) {
      v16bf a0 = *(const v16bf*)(&ldsA[((0 * 16 + kt) << 9) + (lane << 4)]);
      v16bf a1 = *(const v16bf*)(&ldsA[((1 * 16 + kt) << 9) + (lane << 4)]);
#pragma unroll
      for (int j = 0; j < 4; ++j) {
        const unsigned short* bp =
            whB + ((size_t)(((nt0 + j) * 16 + kt)) << 9) + (lane << 4);
        v16bf bf = *(const v16bf*)bp;
        acc[0][j] = wmma_bf16(a0, bf, acc[0][j]);
        acc[1][j] = wmma_bf16(a1, bf, acc[1][j]);
      }
    }

    // gate = sigmoid(h_proj + (c_proj + wh_b)); alpha[m] += qt[n]*gate.
#pragma unroll
    for (int mt = 0; mt < 2; ++mt)
#pragma unroll
      for (int i = 0; i < 8; ++i) {
        int m = (mt << 4) + i + (laneHi << 3);
        float p = 0.f;
#pragma unroll
        for (int j = 0; j < 4; ++j) {
          int n = ((nt0 + j) << 4) + nl;
          float x = acc[mt][j][i] + ldsCW[(m << 9) + n];
          float g = 1.0f / (1.0f + __expf(-x));
          p += qn[j] * g;
        }
        atomicAdd(&alphaS[m], p);  // ds_add_f32
      }
    __syncthreads();

    // summed[m,d] += alpha[m] * hist[m,d]  (hist re-read from LDS, not HBM).
    {
      int m = tid >> 3;
      int d0 = (tid & 7) << 6;
      float al = alphaS[m];
      int mt = m >> 4, ml = m & 15;
      for (int dd = 0; dd < 64; dd += 4) {
        int d = d0 + dd;
        int off = frag_off((mt << 4) + (d >> 5), d & 31, ml);
        ushort4 q = *(const ushort4*)(&ldsA[off]);
        float* sp = &ldsSum[(m << 9) + d];
        sp[0] += al * bf2f(q.x);
        sp[1] += al * bf2f(q.y);
        sp[2] += al * bf2f(q.z);
        sp[3] += al * bf2f(q.w);
      }
    }
  }
  __syncthreads();

  // Flush summed tile to workspace.
  {
    int m = tid >> 3;
    int d0 = (tid & 7) << 6;
    float* dst = summed + ((size_t)b * NS + s0 + m) * D2 + d0;
    for (int dd = 0; dd < 64; dd += 4)
      *(float4*)(dst + dd) = *(const float4*)(&ldsSum[(m << 9) + d0 + dd]);
  }
}

// ---------------------------------------------------------------------------
extern "C" void kernel_launch(void* const* d_in, const int* in_sizes, int n_in,
                              void* d_out, int out_size, void* d_ws,
                              size_t ws_size, hipStream_t stream) {
  (void)in_sizes; (void)n_in; (void)out_size; (void)ws_size;
  const float* hist = (const float*)d_in[0];
  const float* cur  = (const float*)d_in[1];
  const float* wc_w = (const float*)d_in[2];
  const float* wc_b = (const float*)d_in[3];
  const float* wh_w = (const float*)d_in[4];
  const float* wh_b = (const float*)d_in[5];
  const float* qt_w = (const float*)d_in[6];
  const float* qt_b = (const float*)d_in[7];
  const float* wf_w = (const float*)d_in[8];
  const float* wf_b = (const float*)d_in[9];
  float* out = (float*)d_out;

  // Workspace layout (~33.6 MB total).
  char* ws = (char*)d_ws;
  unsigned short* wcB = (unsigned short*)ws; ws += (size_t)D2 * D2 * 2;
  unsigned short* whB = (unsigned short*)ws; ws += (size_t)D2 * D2 * 2;
  unsigned short* wfB = (unsigned short*)ws; ws += (size_t)NE * 2 * D2 * 2;
  float* c_proj = (float*)ws;                ws += (size_t)NB * NS * D2 * 4;
  float* summed = (float*)ws;

  // 0) weight bf16 fragment swizzles
  w_swizzle<<<D2 * D2 / 1024, 256, 0, stream>>>(wc_w, wcB, D2, D2);
  w_swizzle<<<D2 * D2 / 1024, 256, 0, stream>>>(wh_w, whB, D2, D2);
  w_swizzle<<<NE * 2 * D2 / 1024, 256, 0, stream>>>(wf_w, wfB, NE, 2 * D2);

  const int nWG = NB * NS / 32;  // 256 workgroups, 256 threads each

  // 1) c_proj = cur @ wc_w^T + wc_b
  gemm_bf16<4><<<nWG, 256, 32 * D2 * 2, stream>>>(
      cur, cur, D2, D2, wcB, wc_b, c_proj, D2);

  // 2) fused gate/alpha/weighted-sum over history (full LDS budget always,
  //    host-side __has_builtin may disagree with the device pass)
  fused_attn<<<nWG, 256, SMEM_TOTAL, stream>>>(
      hist, c_proj, whB, wh_b, qt_w, qt_b, summed);

  // 3) final = [cur | summed] @ wf_w^T + wf_b
  gemm_bf16<2><<<nWG, 256, 32 * 2 * D2 * 2, stream>>>(
      cur, summed, D2, 2 * D2, wfB, wf_b, out, NE);
}
